// SDM_43671227465846
// MI455X (gfx1250) — compile-verified, matrix operand order checked by reading, not run on hardware
//
#include <hip/hip_runtime.h>
#include <hip/hip_bf16.h>

#define BB 4096
#define LL 50
#define DD 128
#define NHOPS 3

typedef __attribute__((ext_vector_type(16))) __bf16 v16bf;
typedef __attribute__((ext_vector_type(8)))  float  v8f;

// ---------- bf16 <-> f32 via native converts (gfx1250 has bf16 VALU) ----------
static __device__ __forceinline__ __bf16 f2bf(float f) { return (__bf16)f; }
static __device__ __forceinline__ float bf2f(__bf16 b) { return (float)b; }

static __device__ __forceinline__ v8f zero8() {
  v8f z;
#pragma unroll
  for (int i = 0; i < 8; ++i) z[i] = 0.f;
  return z;
}

static __device__ __forceinline__ v8f wmma_bf16(v16bf a, v16bf b, v8f c) {
  // D = A(16x32 bf16) x B(32x16 bf16) + C(16x16 f32)
  return __builtin_amdgcn_wmma_f32_16x16x32_bf16(false, a, false, b, (short)0, c,
                                                 false, false);
}

static __device__ __forceinline__ float wave_sum(float v) {
#pragma unroll
  for (int o = 16; o > 0; o >>= 1) v += __shfl_xor(v, o, 32);
  return v;
}

// ---------- fragment loaders ----------
// Operand fragment layout (16x32 bf16 tile), lane = row/col (lane&15):
//   e=0..7  -> K = k0 + off + e
//   e=8..15 -> K = k0 + off + 16 + (e-8)   where off = (lane&16)?8:0
static __device__ __forceinline__ v16bf load_afrag_f32(const float* __restrict__ rowp,
                                                       int k0) {
  int lane = threadIdx.x & 31;
  int off = (lane & 16) ? 8 : 0;
  const float* p = rowp + k0 + off;
  v16bf r;
#pragma unroll
  for (int e = 0; e < 8; ++e) {
    r[e]     = f2bf(p[e]);
    r[8 + e] = f2bf(p[16 + e]);
  }
  return r;
}

// generic bf16 fragment from LDS (A tile: r0=0,ldk=tile-ld; B tile: r0=n0,ldk=K)
static __device__ __forceinline__ v16bf load_frag_lds(const __bf16* base, int ldk,
                                                      int r0, int k0) {
  int lane = threadIdx.x & 31;
  int rr = r0 + (lane & 15);
  int off = (lane & 16) ? 8 : 0;
  const __bf16* p = base + rr * ldk + k0 + off;
  v16bf r;
#pragma unroll
  for (int e = 0; e < 8; ++e) {
    r[e]     = p[e];
    r[8 + e] = p[16 + e];
  }
  return r;
}

static __device__ __forceinline__ v16bf load_afrag_lds_f32(const float* base, int k0) {
  int lane = threadIdx.x & 31;
  int row = lane & 15;
  int off = (lane & 16) ? 8 : 0;
  const float* p = base + row * DD + k0 + off;
  v16bf r;
#pragma unroll
  for (int e = 0; e < 8; ++e) {
    r[e]     = f2bf(p[e]);
    r[8 + e] = f2bf(p[16 + e]);
  }
  return r;
}

// B fragment from row-major global weight W[N][K] (computing X @ W^T)
static __device__ __forceinline__ v16bf load_bfrag(const __bf16* __restrict__ W,
                                                   int ldk, int n0, int k0) {
  int lane = threadIdx.x & 31;
  int n = n0 + (lane & 15);
  int off = (lane & 16) ? 8 : 0;
  const __bf16* p = W + (size_t)n * ldk + k0 + off;
  v16bf r;
#pragma unroll
  for (int e = 0; e < 8; ++e) {
    r[e]     = p[e];
    r[8 + e] = p[16 + e];
  }
  return r;
}

// ============================================================================
// Kernel 1: convert / split weights to bf16 (W2 split into q-part and m-part)
// ============================================================================
__global__ void sdm_prep_weights(const float* __restrict__ A_W1,
                                 const float* __restrict__ C_W1,
                                 const float* __restrict__ A_W2,
                                 const float* __restrict__ C_W2,
                                 const float* __restrict__ gate_w,
                                 const float* __restrict__ out_w,
                                 __bf16* __restrict__ w1a, __bf16* __restrict__ w1c,
                                 __bf16* __restrict__ w2qa, __bf16* __restrict__ w2ma,
                                 __bf16* __restrict__ w2qc, __bf16* __restrict__ w2mc,
                                 __bf16* __restrict__ gw, __bf16* __restrict__ ow) {
  int t = blockIdx.x * blockDim.x + threadIdx.x;
  int stride = gridDim.x * blockDim.x;
  for (int i = t; i < DD * 2 * DD; i += stride) {
    w1a[i] = f2bf(A_W1[i]);
    w1c[i] = f2bf(C_W1[i]);
  }
  for (int i = t; i < DD * DD; i += stride) {
    int n = i >> 7, k = i & 127;
    w2qa[i] = f2bf(A_W2[n * 256 + k]);
    w2ma[i] = f2bf(A_W2[n * 256 + 128 + k]);
    w2qc[i] = f2bf(C_W2[n * 256 + k]);
    w2mc[i] = f2bf(C_W2[n * 256 + 128 + k]);
    gw[i]   = f2bf(gate_w[i]);
    ow[i]   = f2bf(out_w[i]);
  }
}

// ============================================================================
// Kernel 2: queries. q = tanh([u;j] @ W1^T + b1).  M=4096, K=256, N=128.
// C side (chain=1) immediately chains q_o through W2qC: yC = q_o @ W2qC^T + b2C
// (hop-invariant -> the hop kernel never needs q_o). 4 waves/block, 16 rows/wave.
// ============================================================================
__global__ void __launch_bounds__(128)
sdm_queries(const int* __restrict__ uids, const int* __restrict__ iids,
            const float* __restrict__ Uemb, const float* __restrict__ Iemb,
            const __bf16* __restrict__ W1, const float* __restrict__ b1,
            float* __restrict__ qout,
            const __bf16* __restrict__ W2q, const float* __restrict__ b2,
            float* __restrict__ yC, int chain) {
  __shared__ __align__(16) __bf16 tile[4][16 * DD];
  int wave = threadIdx.x >> 5;
  int lane = threadIdx.x & 31;
  int row0 = blockIdx.x * 64 + wave * 16;
  int rsel = row0 + (lane & 15);
  const float* urow = Uemb + (size_t)uids[rsel] * DD;
  const float* jrow = Iemb + (size_t)iids[rsel] * DD;

  v8f acc[8];
#pragma unroll
  for (int i = 0; i < 8; ++i) acc[i] = zero8();

#pragma unroll
  for (int ks = 0; ks < 8; ++ks) {
    int k0 = ks * 32;
    const float* rp = (k0 < DD) ? urow : (jrow - DD);  // concat gather
    v16bf a = load_afrag_f32(rp, k0);
#pragma unroll
    for (int nt = 0; nt < 8; ++nt) {
      v16bf b = load_bfrag(W1, 2 * DD, nt * 16, k0);
      acc[nt] = wmma_bf16(a, b, acc[nt]);
    }
  }

  int ncol = lane & 15;
  int mbase = (lane & 16) ? 8 : 0;
  __bf16* mytile = &tile[wave][0];
#pragma unroll
  for (int nt = 0; nt < 8; ++nt) {
#pragma unroll
    for (int i = 0; i < 8; ++i) {
      int n = nt * 16 + ncol;
      int m = mbase + i;
      float v = tanhf(acc[nt][i] + b1[n]);
      if (qout) qout[(size_t)(row0 + m) * DD + n] = v;
      mytile[m * DD + n] = f2bf(v);
    }
  }

  if (chain) {
    v8f a2[8];
#pragma unroll
    for (int i = 0; i < 8; ++i) a2[i] = zero8();
#pragma unroll
    for (int ks = 0; ks < 4; ++ks) {
      int k0 = ks * 32;
      v16bf a = load_frag_lds(mytile, DD, 0, k0);
#pragma unroll
      for (int nt = 0; nt < 8; ++nt) {
        v16bf b = load_bfrag(W2q, DD, nt * 16, k0);
        a2[nt] = wmma_bf16(a, b, a2[nt]);
      }
    }
#pragma unroll
    for (int nt = 0; nt < 8; ++nt)
#pragma unroll
      for (int i = 0; i < 8; ++i) {
        int n = nt * 16 + ncol;
        int m = mbase + i;
        yC[(size_t)(row0 + m) * DD + n] = a2[nt][i] + b2[n];
      }
  }
}

// ============================================================================
// Kernel 3: Ma[m,:] = A_item_emb[seq[m]] @ W2mA^T + b2A   (bf16 out)
// Weights cached in LDS (32 KB) once per block; B-frags come from ds loads.
// ============================================================================
__global__ void __launch_bounds__(128)
sdm_build_ma(const int* __restrict__ seqs, const float* __restrict__ IembA,
             const __bf16* __restrict__ W2m, const float* __restrict__ b2,
             __bf16* __restrict__ Ma) {
  __shared__ __align__(16) __bf16 wlds[DD * DD];
  int tid = threadIdx.x;
  {
    const uint4* src = (const uint4*)W2m;
    uint4* dst = (uint4*)wlds;
    for (int i = tid; i < DD * DD / 8; i += 128) dst[i] = src[i];
  }
  __syncthreads();

  int wave = tid >> 5;
  int lane = tid & 31;
  int row0 = blockIdx.x * 64 + wave * 16;
  int rsel = row0 + (lane & 15);
  const float* rp = IembA + (size_t)seqs[rsel] * DD;

  v8f acc[8];
#pragma unroll
  for (int i = 0; i < 8; ++i) acc[i] = zero8();
#pragma unroll
  for (int ks = 0; ks < 4; ++ks) {
    int k0 = ks * 32;
    v16bf a = load_afrag_f32(rp, k0);
#pragma unroll
    for (int nt = 0; nt < 8; ++nt)
      acc[nt] = wmma_bf16(a, load_frag_lds(wlds, DD, nt * 16, k0), acc[nt]);
  }
  int ncol = lane & 15;
  int mbase = (lane & 16) ? 8 : 0;
#pragma unroll
  for (int nt = 0; nt < 8; ++nt)
#pragma unroll
    for (int i = 0; i < 8; ++i) {
      int n = nt * 16 + ncol;
      int m = mbase + i;
      Ma[(size_t)(row0 + m) * DD + n] = f2bf(acc[nt][i] + b2[n]);
    }
}

// ============================================================================
// Kernel 4: P[m,:] = tanh(yC[b] + m_c@W2mC^T)^2 @ out_w^T   (bf16 out)
// First GEMM's weights cached in LDS; chained second GEMM via LDS bf16 tile.
// ============================================================================
__global__ void __launch_bounds__(128)
sdm_build_p(const int* __restrict__ seqs, const float* __restrict__ IembC,
            const __bf16* __restrict__ W2m, const float* __restrict__ yC,
            const __bf16* __restrict__ outW, __bf16* __restrict__ P) {
  __shared__ __align__(16) __bf16 wlds[DD * DD];
  __shared__ __align__(16) __bf16 tile[4][16 * DD];
  int tid = threadIdx.x;
  {
    const uint4* src = (const uint4*)W2m;
    uint4* dst = (uint4*)wlds;
    for (int i = tid; i < DD * DD / 8; i += 128) dst[i] = src[i];
  }
  __syncthreads();

  int wave = tid >> 5;
  int lane = tid & 31;
  int row0 = blockIdx.x * 64 + wave * 16;
  int rsel = row0 + (lane & 15);
  const float* rp = IembC + (size_t)seqs[rsel] * DD;

  v8f acc[8];
#pragma unroll
  for (int i = 0; i < 8; ++i) acc[i] = zero8();
#pragma unroll
  for (int ks = 0; ks < 4; ++ks) {
    int k0 = ks * 32;
    v16bf a = load_afrag_f32(rp, k0);
#pragma unroll
    for (int nt = 0; nt < 8; ++nt)
      acc[nt] = wmma_bf16(a, load_frag_lds(wlds, DD, nt * 16, k0), acc[nt]);
  }

  int ncol = lane & 15;
  int mbase = (lane & 16) ? 8 : 0;
  __bf16* mytile = &tile[wave][0];
#pragma unroll
  for (int nt = 0; nt < 8; ++nt)
#pragma unroll
    for (int i = 0; i < 8; ++i) {
      int n = nt * 16 + ncol;
      int m = mbase + i;
      int mg = row0 + m;
      int bidx = mg / LL;
      float v = tanhf(acc[nt][i] + yC[(size_t)bidx * DD + n]);
      mytile[m * DD + n] = f2bf(v * v);
    }

  v8f a2[8];
#pragma unroll
  for (int i = 0; i < 8; ++i) a2[i] = zero8();
#pragma unroll
  for (int ks = 0; ks < 4; ++ks) {
    int k0 = ks * 32;
    v16bf a = load_frag_lds(mytile, DD, 0, k0);
#pragma unroll
    for (int nt = 0; nt < 8; ++nt)
      a2[nt] = wmma_bf16(a, load_bfrag(outW, DD, nt * 16, k0), a2[nt]);
  }
#pragma unroll
  for (int nt = 0; nt < 8; ++nt)
#pragma unroll
    for (int i = 0; i < 8; ++i) {
      int n = nt * 16 + ncol;
      int m = mbase + i;
      P[(size_t)(row0 + m) * DD + n] = f2bf(a2[nt][i]);
    }
}

// ============================================================================
// Kernel 5: fused 3-hop loop. 256 threads (8 waves) own 16 batch rows.
// Weight B-fragments (W2qA, gate_w) are hop-invariant -> hoisted into VGPRs.
// ============================================================================
__global__ void __launch_bounds__(256)
sdm_hops(const int* __restrict__ iids, const float* __restrict__ mask,
         const float* __restrict__ q0, const __bf16* __restrict__ Ma,
         const __bf16* __restrict__ P, const __bf16* __restrict__ W2qA,
         const __bf16* __restrict__ gateW, const float* __restrict__ gate_b,
         const float* __restrict__ out_b, const float* __restrict__ sum_w,
         const float* __restrict__ sum_b, const float* __restrict__ item_bias,
         float* __restrict__ out) {
  __shared__ float qs[16 * DD];
  __shared__ float yAs[16 * DD];
  __shared__ float gls[16 * DD];
  __shared__ float sc[16 * LL];

  int tid = threadIdx.x;
  int wave = tid >> 5;
  int lane = tid & 31;
  int rowbase = blockIdx.x * 16;

  for (int i = tid; i < 16 * DD; i += 256) qs[i] = q0[(size_t)rowbase * DD + i];

  // hop-invariant weight fragments: wave w owns output tile n = w*16..w*16+15
  v16bf wby[4], wbg[4];
#pragma unroll
  for (int ks = 0; ks < 4; ++ks) {
    wby[ks] = load_bfrag(W2qA, DD, wave * 16, ks * 32);
    wbg[ks] = load_bfrag(gateW, DD, wave * 16, ks * 32);
  }
  __syncthreads();

  for (int hop = 0; hop < NHOPS; ++hop) {
    // (a) yA = q @ W2qA^T (b2A folded into Ma), glog = q @ gate_w^T
    v8f ay = zero8(), ag = zero8();
#pragma unroll
    for (int ks = 0; ks < 4; ++ks) {
      v16bf a = load_afrag_lds_f32(qs, ks * 32);
      ay = wmma_bf16(a, wby[ks], ay);
      ag = wmma_bf16(a, wbg[ks], ag);
    }
    {
      int n = wave * 16 + (lane & 15);
      int mbase = (lane & 16) ? 8 : 0;
#pragma unroll
      for (int i = 0; i < 8; ++i) {
        yAs[(mbase + i) * DD + n] = ay[i];
        gls[(mbase + i) * DD + n] = ag[i];
      }
    }
    __syncthreads();

    // (b) scores: s[r,l] = -sum_d tanh(yA[r,d] + Ma[b,l,d])^2 * mask
    for (int task = wave; task < 16 * LL; task += 8) {
      int r = task & 15;
      int l = task >> 4;
      size_t base = (((size_t)(rowbase + r)) * LL + l) * DD + lane * 4;
      float s = 0.f;
#pragma unroll
      for (int j = 0; j < 4; ++j) {
        float t = tanhf(yAs[r * DD + lane * 4 + j] + bf2f(Ma[base + j]));
        s += t * t;
      }
      s = wave_sum(s);
      if (lane == 0) sc[r * LL + l] = -s * mask[(size_t)(rowbase + r) * LL + l];
    }
    __syncthreads();

    // (c) softmax over L=50 per row; wave w handles rows w and w+8
    for (int r = wave; r < 16; r += 8) {
      float s0 = (lane < LL) ? sc[r * LL + lane] : -1e30f;
      float s1 = (lane + 32 < LL) ? sc[r * LL + lane + 32] : -1e30f;
      float mx = fmaxf(s0, s1);
#pragma unroll
      for (int o = 16; o > 0; o >>= 1) mx = fmaxf(mx, __shfl_xor(mx, o, 32));
      float e0 = (lane < LL) ? __expf(s0 - mx) : 0.f;
      float e1 = (lane + 32 < LL) ? __expf(s1 - mx) : 0.f;
      float sm = wave_sum(e0 + e1);
      float inv = 1.f / sm;
      if (lane < LL) sc[r * LL + lane] = e0 * inv;
      if (lane + 32 < LL) sc[r * LL + lane + 32] = e1 * inv;
    }
    __syncthreads();

    // (d) o = -sum_l relu(w*P + out_b); g = sigmoid(glog + gate_b); q update
    {
      int d = tid & 127;
      int half = tid >> 7;
      float ob = out_b[d];
      float gb = gate_b[d];
      for (int r = half; r < 16; r += 2) {
        size_t base = (((size_t)(rowbase + r)) * LL) * DD + d;
        float o = 0.f;
#pragma unroll 5
        for (int l = 0; l < LL; ++l) {
          float w = sc[r * LL + l];
          o += fmaxf(w * bf2f(P[base + (size_t)l * DD]) + ob, 0.f);
        }
        o = -o;
        float g = 1.f / (1.f + __expf(-(gls[r * DD + d] + gb)));
        qs[r * DD + d] = g * qs[r * DD + d] + (1.f - g) * o;
      }
    }
    __syncthreads();
  }

  // final: out[b] = q[b] . sum_w + sum_b + C_item_bias[item_ids[b]]
  for (int r = wave; r < 16; r += 8) {
    float dsum = 0.f;
#pragma unroll
    for (int j = 0; j < 4; ++j)
      dsum += qs[r * DD + lane * 4 + j] * sum_w[lane * 4 + j];
    dsum = wave_sum(dsum);
    if (lane == 0) {
      int gr = rowbase + r;
      out[gr] = dsum + sum_b[0] + item_bias[iids[gr]];
    }
  }
}

// ============================================================================
extern "C" void kernel_launch(void* const* d_in, const int* in_sizes, int n_in,
                              void* d_out, int out_size, void* d_ws, size_t ws_size,
                              hipStream_t stream) {
  (void)in_sizes; (void)n_in; (void)out_size; (void)ws_size;
  const int*   user_ids    = (const int*)d_in[0];
  const int*   item_ids    = (const int*)d_in[1];
  const int*   item_seqs   = (const int*)d_in[2];
  const float* mask        = (const float*)d_in[3];
  const float* A_user_emb  = (const float*)d_in[4];
  const float* A_item_emb  = (const float*)d_in[5];
  const float* C_user_emb  = (const float*)d_in[6];
  const float* C_item_emb  = (const float*)d_in[7];
  const float* C_item_bias = (const float*)d_in[8];
  const float* A_W1_w = (const float*)d_in[9];
  const float* A_W1_b = (const float*)d_in[10];
  const float* A_W2_w = (const float*)d_in[11];
  const float* A_W2_b = (const float*)d_in[12];
  const float* C_W1_w = (const float*)d_in[13];
  const float* C_W1_b = (const float*)d_in[14];
  const float* C_W2_w = (const float*)d_in[15];
  const float* C_W2_b = (const float*)d_in[16];
  const float* gate_w = (const float*)d_in[17];
  const float* gate_b = (const float*)d_in[18];
  const float* out_w  = (const float*)d_in[19];
  const float* out_b  = (const float*)d_in[20];
  const float* sum_w  = (const float*)d_in[21];
  const float* sum_b  = (const float*)d_in[22];

  char* ws = (char*)d_ws;
  __bf16* w1a  = (__bf16*)(ws + 0);        // 128x256
  __bf16* w1c  = (__bf16*)(ws + 65536);    // 128x256
  __bf16* w2qa = (__bf16*)(ws + 131072);   // 128x128 each below
  __bf16* w2ma = (__bf16*)(ws + 163840);
  __bf16* w2qc = (__bf16*)(ws + 196608);
  __bf16* w2mc = (__bf16*)(ws + 229376);
  __bf16* gw   = (__bf16*)(ws + 262144);
  __bf16* ow   = (__bf16*)(ws + 294912);
  float*  q0   = (float*)(ws + 393216);                       // [4096][128] f32
  float*  yC   = (float*)(ws + 393216 + 2097152);             // [4096][128] f32
  __bf16* Ma   = (__bf16*)(ws + 393216 + 2 * 2097152);        // [204800][128] bf16
  __bf16* P    = (__bf16*)(ws + 393216 + 2 * 2097152 + 52428800);

  sdm_prep_weights<<<64, 256, 0, stream>>>(A_W1_w, C_W1_w, A_W2_w, C_W2_w, gate_w,
                                           out_w, w1a, w1c, w2qa, w2ma, w2qc, w2mc,
                                           gw, ow);
  // A-side query: q0 = tanh([uA;jA] @ A_W1^T + b)
  sdm_queries<<<64, 128, 0, stream>>>(user_ids, item_ids, A_user_emb, A_item_emb,
                                      w1a, A_W1_b, q0, nullptr, nullptr, nullptr, 0);
  // C-side query chained: yC = tanh([uC;jC] @ C_W1^T + b) @ W2qC^T + b2C
  sdm_queries<<<64, 128, 0, stream>>>(user_ids, item_ids, C_user_emb, C_item_emb,
                                      w1c, C_W1_b, nullptr, w2qc, C_W2_b, yC, 1);
  // hop-invariant memories
  sdm_build_ma<<<3200, 128, 0, stream>>>(item_seqs, A_item_emb, w2ma, A_W2_b, Ma);
  sdm_build_p<<<3200, 128, 0, stream>>>(item_seqs, C_item_emb, w2mc, yC, ow, P);
  // fused 3-hop loop + final scoring
  sdm_hops<<<256, 256, 0, stream>>>(item_ids, mask, q0, Ma, P, w2qa, gw,
                                    gate_b, out_b, sum_w, sum_b, C_item_bias,
                                    (float*)d_out);
}
